// SingleTargetPose_29240137351329
// MI455X (gfx1250) — matrix operand order repaired and checked by reading, not compiled, not run on hardware
//
#include <hip/hip_runtime.h>
#include <hip/hip_bf16.h>
#include <hip/hip_fp16.h>

typedef __attribute__((ext_vector_type(16))) _Float16 v16h;
typedef __attribute__((ext_vector_type(8)))  _Float16 v8h;
typedef __attribute__((ext_vector_type(8)))  float    v8f;

struct H16 { v8h lo, hi; };

#define BN_EPS 1e-3f

__device__ __forceinline__ float sigmoidf_(float x) { return 1.f / (1.f + __expf(-x)); }

// Per-lane async DMA: global (16B) -> LDS, tracked by ASYNCcnt.
__device__ __forceinline__ void async_g2l_b128(unsigned lds_off, const void* gptr) {
    asm volatile("global_load_async_to_lds_b128 %0, %1, off"
                 :: "v"(lds_off), "v"((unsigned long long)(uintptr_t)gptr)
                 : "memory");
}
__device__ __forceinline__ void wait_async0() {
    asm volatile("s_wait_asynccnt 0x0" ::: "memory");
}
// Generic shared-space pointer -> 32-bit LDS offset (aperture puts offset in addr[31:0])
__device__ __forceinline__ unsigned lds_off32(const void* p) {
    return (unsigned)(unsigned long long)(uintptr_t)p;
}

// ---------------------------------------------------------------------------
// NCHW f32 -> NHWC f16
// ---------------------------------------------------------------------------
__global__ void nchw_to_nhwc_f16(const float* __restrict__ Xin, _Float16* __restrict__ Xh,
                                 int C, int HW, int Ptot) {
    int gid = blockIdx.x * 256 + threadIdx.x;
    if (gid >= Ptot * C) return;
    int p = gid / C, ch = gid - p * C;
    int b = p / HW, hw = p - b * HW;
    Xh[gid] = (_Float16)Xin[((size_t)b * C + ch) * HW + hw];
}

// ---------------------------------------------------------------------------
// Pack conv weights (O,I,kh,kw) f32 -> transposed GEMM form [Opad][taps*I] f16
// K ordering: kk = (ky*3+kx)*I + ci
// ---------------------------------------------------------------------------
__global__ void pack_wt(const float* __restrict__ w, _Float16* __restrict__ wt,
                        int O, int I, int taps, int Opad) {
    int K = taps * I;
    int gid = blockIdx.x * 256 + threadIdx.x;
    if (gid >= Opad * K) return;
    int o = gid / K, kk = gid - o * K;
    int tap = kk / I, ci = kk - tap * I;
    float v = 0.f;
    if (o < O) v = w[((size_t)o * I + ci) * taps + tap];
    wt[gid] = (_Float16)v;
}

// Fold BN into per-channel scale/bias (or plain conv bias: scale=1)
__global__ void pack_sb(const float* __restrict__ g, const float* __restrict__ be,
                        const float* __restrict__ me, const float* __restrict__ va,
                        const float* __restrict__ bc, float* __restrict__ sc,
                        float* __restrict__ bi, int O, int Opad, int bn) {
    int o = blockIdx.x * 64 + threadIdx.x;
    if (o >= Opad) return;
    if (o >= O) { sc[o] = 0.f; bi[o] = 0.f; return; }
    if (bn) {
        float s = g[o] * rsqrtf(va[o] + BN_EPS);
        sc[o] = s; bi[o] = be[o] - me[o] * s;
    } else {
        sc[o] = 1.f; bi[o] = bc[o];
    }
}

// ---------------------------------------------------------------------------
// Implicit-GEMM conv via V_WMMA_F32_16X16X32_F16, double-buffered async
// Global->LDS staging (DMA of chunk k+1 overlaps WMMA of chunk k).
//   X : NHWC f16 [Ptot][Cin], Wt : [Opad][K] f16 (K-contiguous per out channel)
//   block = 256 threads (8 waves); tile M=128 pixels, N=64 channels, K-chunk=64
//   mode 0: BN+SiLU -> f16 [Ptot][CoutStore]
//   mode 1: +bias   -> f32 Yf[b*bstride + (coff+n)*8400 + aoff + a_local], n<CoutStore
// ---------------------------------------------------------------------------
__global__ __launch_bounds__(256)
void conv_wmma(const _Float16* __restrict__ X, const _Float16* __restrict__ Wt,
               const float* __restrict__ scale, const float* __restrict__ bias,
               _Float16* __restrict__ Yh, float* __restrict__ Yf,
               int H, int W, int Cin, int taps, int CoutStore,
               int HW, int Ptot, int bstride, int coff, int aoff, int mode) {
    __shared__ _Float16 As[2][128][64];   // 2 x 16 KB
    __shared__ _Float16 Bs[2][64][64];    // 2 x  8 KB   (Bs[buf][n][k])
    const int tid  = threadIdx.x;
    const int wave = tid >> 5;
    const int lane = tid & 31;
    const int m0 = blockIdx.x * 128;
    const int n0 = blockIdx.y * 64;
    const int K  = taps * Cin;
    const unsigned As_base = lds_off32(&As[0][0][0]);
    const unsigned Bs_base = lds_off32(&Bs[0][0][0]);
    v8f acc[4] = {};

    // ---- per-thread loop-invariant staging descriptors ----
    int    a_r[4], a_g[4], a_y[4], a_x[4];
    bool   a_pv[4];
    size_t a_bhw[4];
    #pragma unroll
    for (int t4 = 0; t4 < 4; ++t4) {
        int task = t4 * 256 + tid;
        int r = task >> 3, g = task & 7;
        int p = m0 + r;
        int b  = p / HW;
        int hw = p - b * HW;
        int q  = hw / W;
        a_r[t4] = r; a_g[t4] = g;
        a_y[t4] = q; a_x[t4] = hw - q * W;
        a_bhw[t4] = (size_t)b * HW;
        a_pv[t4] = (p < Ptot);
    }
    int b_n[2], b_g[2];
    const _Float16* b_row[2];
    #pragma unroll
    for (int t2 = 0; t2 < 2; ++t2) {
        int task = t2 * 256 + tid;
        int n = task >> 3, g = task & 7;
        b_n[t2] = n; b_g[t2] = g;
        b_row[t2] = Wt + (size_t)(n0 + n) * K + g * 8;
    }

    // ---- stage one K-chunk (64 wide) into buffer bufi ----
    auto stage = [&](int k0, int bufi) {
        int ci0 = k0, dy = 0, dx = 0;
        if (taps == 9) {
            int tap = k0 / Cin;          // 64-chunk never straddles taps (Cin % 64 == 0)
            ci0 = k0 - tap * Cin;
            dy = tap / 3 - 1; dx = tap % 3 - 1;
        }
        #pragma unroll
        for (int t4 = 0; t4 < 4; ++t4) {
            int yy = a_y[t4] + dy, xx = a_x[t4] + dx;
            unsigned loff = As_base +
                (unsigned)(bufi * (128 * 64) + a_r[t4] * 64 + a_g[t4] * 8) * 2u;
            if (a_pv[t4] && yy >= 0 && yy < H && xx >= 0 && xx < W) {
                const _Float16* src =
                    X + (a_bhw[t4] + (size_t)yy * W + xx) * Cin + ci0 + a_g[t4] * 8;
                async_g2l_b128(loff, src);
            } else {
                *(v8h*)&As[bufi][a_r[t4]][a_g[t4] * 8] = (v8h){};   // zero-pad halo
            }
        }
        #pragma unroll
        for (int t2 = 0; t2 < 2; ++t2) {
            async_g2l_b128(Bs_base +
                (unsigned)(bufi * (64 * 64) + b_n[t2] * 64 + b_g[t2] * 8) * 2u,
                b_row[t2] + k0);
        }
    };

    const int nk = K >> 6;
    stage(0, 0);
    const int row  = (wave << 4) + (lane & 15);
    const int klo  = (lane & 16) ? 8 : 0;
    const int kb   = (lane & 16) ? 8 : 0;
    const int colb = lane & 15;

    for (int kc = 0; kc < nk; ++kc) {
        const int bufi = kc & 1;
        wait_async0();              // own-wave DMAs for chunk kc done
        __syncthreads();            // all waves published; prior buffer reads finished
        if (kc + 1 < nk) stage((kc + 1) << 6, bufi ^ 1);   // overlap DMA with WMMA

        #pragma unroll
        for (int s = 0; s < 2; ++s) {
            const int ko = s * 32;
            H16 ha;
            ha.lo = *(const v8h*)&As[bufi][row][ko + klo];
            ha.hi = *(const v8h*)&As[bufi][row][ko + klo + 16];
            v16h a = __builtin_bit_cast(v16h, ha);
            #pragma unroll
            for (int t = 0; t < 4; ++t) {
                H16 hb;
                hb.lo = *(const v8h*)&Bs[bufi][colb + t * 16][ko + kb];
                hb.hi = *(const v8h*)&Bs[bufi][colb + t * 16][ko + kb + 16];
                v16h bf = __builtin_bit_cast(v16h, hb);
                acc[t] = __builtin_amdgcn_wmma_f32_16x16x32_f16(
                    false, a, false, bf, (short)0, acc[t], false, false);
            }
        }
    }

    // ---- epilogue ----
    const int ncol = lane & 15;
    const int mb   = (wave << 4) + ((lane & 16) ? 8 : 0);
    #pragma unroll
    for (int t = 0; t < 4; ++t) {
        int n = n0 + t * 16 + ncol;
        float sc = scale[n], bi = bias[n];
        #pragma unroll
        for (int r = 0; r < 8; ++r) {
            int m = m0 + mb + r;
            if (m >= Ptot) continue;
            float yv = acc[t][r] * sc + bi;
            if (mode == 0) {
                float s = yv * sigmoidf_(yv);          // SiLU
                Yh[(size_t)m * CoutStore + n] = (_Float16)s;
            } else if (n < CoutStore) {
                int b  = m / HW;
                int al = m - b * HW;
                Yf[(size_t)b * bstride + (size_t)(coff + n) * 8400 + aoff + al] = yv;
            }
        }
    }
}

// ---------------------------------------------------------------------------
// Depthwise 3x3 CBS (conv + BN + SiLU), NHWC f16
// ---------------------------------------------------------------------------
__global__ void dw_cbs(const _Float16* __restrict__ Xh, const float* __restrict__ w,
                       const float* __restrict__ g, const float* __restrict__ be,
                       const float* __restrict__ me, const float* __restrict__ va,
                       _Float16* __restrict__ Y, int C, int H, int W, int HW, int Ptot) {
    int gid = blockIdx.x * 256 + threadIdx.x;
    if (gid >= Ptot * C) return;
    int p = gid / C, ch = gid - p * C;
    int b = p / HW, hw = p - b * HW;
    int q = hw / W;
    int y = q, x = hw - q * W;
    float acc = 0.f;
    #pragma unroll
    for (int ky = 0; ky < 3; ++ky)
        #pragma unroll
        for (int kx = 0; kx < 3; ++kx) {
            int yy = y + ky - 1, xx = x + kx - 1;
            if (yy < 0 || yy >= H || xx < 0 || xx >= W) continue;
            acc += (float)Xh[((size_t)b * HW + (size_t)yy * W + xx) * C + ch] *
                   w[ch * 9 + ky * 3 + kx];
        }
    float s  = g[ch] * rsqrtf(va[ch] + BN_EPS);
    float yv = (acc - me[ch]) * s + be[ch];
    Y[gid] = (_Float16)(yv * sigmoidf_(yv));
}

// ---------------------------------------------------------------------------
// cls head: 1x1 conv 128->1 (+bias), raw logit into det channel 64
// ---------------------------------------------------------------------------
__global__ void cls_head(const _Float16* __restrict__ T, const float* __restrict__ w,
                         const float* __restrict__ bcv, float* __restrict__ det,
                         int HW, int Ptot, int aoff) {
    int p = blockIdx.x * 256 + threadIdx.x;
    if (p >= Ptot) return;
    float acc = bcv[0];
    const _Float16* row = T + (size_t)p * 128;
    #pragma unroll 8
    for (int i = 0; i < 128; ++i) acc += (float)row[i] * w[i];
    int b = p / HW, al = p - b * HW;
    det[(size_t)b * (65 * 8400) + (size_t)64 * 8400 + aoff + al] = acc;
}

// ---------------------------------------------------------------------------
// Pass 1: per-image reduction over 8400 anchors -> selected index
// ---------------------------------------------------------------------------
__global__ __launch_bounds__(256)
void select_best(const float* __restrict__ det, const float* __restrict__ kpt,
                 int* __restrict__ sel) {
    __shared__ float s_sum[256], s_sq[256], s_mc[256], s_mv[256];
    __shared__ int   s_ic[256], s_iv[256];
    int b = blockIdx.x, t = threadIdx.x;
    const float* db = det + (size_t)b * (65 * 8400) + (size_t)64 * 8400;
    const float* kb = kpt + (size_t)b * (63 * 8400);
    float sum = 0.f, sq = 0.f, mc = -1e30f, mv = -1e30f;
    int ic = 0, iv = 0;
    for (int a = t; a < 8400; a += 256) {
        float cs = sigmoidf_(db[a]);
        float vs = 0.f;
        #pragma unroll
        for (int k = 0; k < 21; ++k) vs += sigmoidf_(kb[(size_t)(3 * k + 2) * 8400 + a]);
        float vm   = vs * (1.f / 21.f);
        float vmc  = fminf(fmaxf(vm, 0.f), 1.f);
        float comb = cs * sqrtf(vmc);
        sum += comb; sq += comb * comb;
        if (comb > mc) { mc = comb; ic = a; }
        if (vmc  > mv) { mv = vmc;  iv = a; }
    }
    s_sum[t] = sum; s_sq[t] = sq;
    s_mc[t] = mc; s_ic[t] = ic; s_mv[t] = mv; s_iv[t] = iv;
    __syncthreads();
    for (int off = 128; off > 0; off >>= 1) {
        if (t < off) {
            s_sum[t] += s_sum[t + off]; s_sq[t] += s_sq[t + off];
            if (s_mc[t + off] > s_mc[t] ||
                (s_mc[t + off] == s_mc[t] && s_ic[t + off] < s_ic[t])) {
                s_mc[t] = s_mc[t + off]; s_ic[t] = s_ic[t + off];
            }
            if (s_mv[t + off] > s_mv[t] ||
                (s_mv[t + off] == s_mv[t] && s_iv[t + off] < s_iv[t])) {
                s_mv[t] = s_mv[t + off]; s_iv[t] = s_iv[t + off];
            }
        }
        __syncthreads();
    }
    if (t == 0) {
        float n   = 8400.f;
        float var = (s_sq[0] - s_sum[0] * s_sum[0] / n) / (n - 1.f);
        float sd  = sqrtf(fmaxf(var, 0.f));
        bool fb   = (s_mc[0] <= 0.001f) || (sd <= 1e-6f);
        sel[b] = fb ? s_iv[0] : s_ic[0];
    }
}

// ---------------------------------------------------------------------------
// Pass 2: decode full det+kpt for the selected anchor, 68 outputs per image
// ---------------------------------------------------------------------------
__global__ void emit_out(const float* __restrict__ det, const float* __restrict__ kpt,
                         const int* __restrict__ sel, float* __restrict__ out) {
    int b = blockIdx.x, t = threadIdx.x;
    if (t >= 68) return;
    int a = sel[b];
    int Wl; float st; int al;
    if (a < 6400)      { Wl = 80; st = 8.f;  al = a; }
    else if (a < 8000) { Wl = 40; st = 16.f; al = a - 6400; }
    else               { Wl = 20; st = 32.f; al = a - 8000; }
    float ax = (float)(al % Wl) + 0.5f;
    float ay = (float)(al / Wl) + 0.5f;
    const float* db = det + (size_t)b * (65 * 8400);
    const float* kb = kpt + (size_t)b * (63 * 8400);
    float r;
    if (t < 4) {
        float dist[4];
        #pragma unroll
        for (int gdx = 0; gdx < 4; ++gdx) {
            float mx = -1e30f;
            for (int k = 0; k < 16; ++k)
                mx = fmaxf(mx, db[(size_t)(gdx * 16 + k) * 8400 + a]);
            float den = 0.f, num = 0.f;
            for (int k = 0; k < 16; ++k) {
                float e = __expf(db[(size_t)(gdx * 16 + k) * 8400 + a] - mx);
                den += e; num += e * (float)k;
            }
            dist[gdx] = num / den;
        }
        float x1 = ax - dist[0], y1 = ay - dist[1];
        float x2 = ax + dist[2], y2 = ay + dist[3];
        float v4[4] = {(x1 + x2) * 0.5f * st, (y1 + y2) * 0.5f * st,
                       (x2 - x1) * st, (y2 - y1) * st};
        r = v4[t];
    } else if (t == 4) {
        r = sigmoidf_(db[(size_t)64 * 8400 + a]);
    } else {
        int j = t - 5, k = j / 3, cm = j - 3 * k;
        float raw = kb[(size_t)(3 * k + cm) * 8400 + a];
        if (cm == 0)      r = (raw * 2.f + (ax - 0.5f)) * st;
        else if (cm == 1) r = (raw * 2.f + (ay - 0.5f)) * st;
        else              r = sigmoidf_(raw);
    }
    out[b * 68 + t] = r;
}

// ---------------------------------------------------------------------------
// Host orchestration
// ---------------------------------------------------------------------------
extern "C" void kernel_launch(void* const* d_in, const int* in_sizes, int n_in,
                              void* d_out, int out_size, void* d_ws, size_t ws_size,
                              hipStream_t stream) {
    (void)out_size; (void)ws_size;
    const bool inputs_first = (in_sizes[0] == 16 * 256 * 80 * 80);
    const int xbase = inputs_first ? 0 : (n_in - 3);
    int cur = inputs_first ? 3 : 0;
    const float* Xin[3] = { (const float*)d_in[xbase + 0],
                            (const float*)d_in[xbase + 1],
                            (const float*)d_in[xbase + 2] };

    struct CBSH { const float *w, *g, *be, *me, *va; };
    struct CVH  { const float *w, *b; };
    auto rdCBS = [&](int& cc) -> CBSH {
        CBSH r;
        if (in_sizes[cc] > in_sizes[cc + 1]) {   // insertion order: w,gamma,beta,mean,var
            r.w  = (const float*)d_in[cc];     r.g  = (const float*)d_in[cc + 1];
            r.be = (const float*)d_in[cc + 2]; r.me = (const float*)d_in[cc + 3];
            r.va = (const float*)d_in[cc + 4];
        } else {                                 // sorted keys: beta,gamma,mean,var,w
            r.be = (const float*)d_in[cc];     r.g  = (const float*)d_in[cc + 1];
            r.me = (const float*)d_in[cc + 2]; r.va = (const float*)d_in[cc + 3];
            r.w  = (const float*)d_in[cc + 4];
        }
        cc += 5; return r;
    };
    auto rdCV = [&](int& cc) -> CVH {
        CVH r;
        if (in_sizes[cc] > in_sizes[cc + 1]) { r.w = (const float*)d_in[cc]; r.b = (const float*)d_in[cc + 1]; }
        else                                 { r.b = (const float*)d_in[cc]; r.w = (const float*)d_in[cc + 1]; }
        cc += 2; return r;
    };

    CBSH cv2a[3], cv2b[3]; CVH cv2c[3];
    for (int i = 0; i < 3; ++i) { cv2a[i] = rdCBS(cur); cv2b[i] = rdCBS(cur); cv2c[i] = rdCV(cur); }
    CBSH cv3a[3], cv3b[3]; CVH cv3c[3];
    for (int i = 0; i < 3; ++i) { cv3a[i] = rdCBS(cur); cv3b[i] = rdCBS(cur); cv3c[i] = rdCV(cur); }
    CBSH cv4a[3], cv4b[3], cv4c3[3]; CVH cv4d[3];
    for (int i = 0; i < 3; ++i) { cv4a[i] = rdCBS(cur); cv4b[i] = rdCBS(cur); cv4c3[i] = rdCBS(cur); cv4d[i] = rdCV(cur); }

    // ---- workspace carve ----
    char* ws = (char*)d_ws;
    auto aln   = [](size_t v) { return (v + 255) & ~(size_t)255; };
    size_t off = 0;
    auto carve = [&](size_t bytes) { size_t o = off; off = aln(off + bytes); return o; };
    size_t OFF_XH  = carve((size_t)26214400 * 2);       // NHWC f16 input  (max: lvl0)
    size_t OFF_DW  = carve((size_t)26214400 * 2);       // depthwise out   (max: lvl0)
    size_t OFF_T1  = carve((size_t)13107200 * 2);       // 128-ch temp
    size_t OFF_T2  = carve((size_t)13107200 * 2);       // 128-ch temp
    size_t OFF_DET = carve((size_t)16 * 65 * 8400 * 4);
    size_t OFF_KPT = carve((size_t)16 * 63 * 8400 * 4);
    size_t OFF_SEL = carve(64 * sizeof(int));
    size_t OFF_SB  = carve(65536);
    size_t OFF_WP  = off;                               // packed weights region

    size_t wp_cur = OFF_WP, sb_cur = OFF_SB;
    struct PKH { const _Float16* wt; const float* sc; const float* bi; int Opad; };
    auto packBN = [&](const CBSH& p, int O, int I, int taps, int Opad) -> PKH {
        _Float16* wt = (_Float16*)(ws + wp_cur);
        wp_cur = aln(wp_cur + (size_t)Opad * taps * I * 2);
        float* sc = (float*)(ws + sb_cur); sb_cur += (size_t)Opad * 4;
        float* bi = (float*)(ws + sb_cur); sb_cur += (size_t)Opad * 4;
        int total = Opad * taps * I;
        pack_wt<<<(total + 255) / 256, 256, 0, stream>>>(p.w, wt, O, I, taps, Opad);
        pack_sb<<<(Opad + 63) / 64, 64, 0, stream>>>(p.g, p.be, p.me, p.va,
                                                     (const float*)nullptr, sc, bi, O, Opad, 1);
        return {wt, sc, bi, Opad};
    };
    auto packBias = [&](const CVH& p, int O, int I, int taps, int Opad) -> PKH {
        _Float16* wt = (_Float16*)(ws + wp_cur);
        wp_cur = aln(wp_cur + (size_t)Opad * taps * I * 2);
        float* sc = (float*)(ws + sb_cur); sb_cur += (size_t)Opad * 4;
        float* bi = (float*)(ws + sb_cur); sb_cur += (size_t)Opad * 4;
        int total = Opad * taps * I;
        pack_wt<<<(total + 255) / 256, 256, 0, stream>>>(p.w, wt, O, I, taps, Opad);
        pack_sb<<<(Opad + 63) / 64, 64, 0, stream>>>((const float*)nullptr, (const float*)nullptr,
                                                     (const float*)nullptr, (const float*)nullptr,
                                                     p.b, sc, bi, O, Opad, 0);
        return {wt, sc, bi, Opad};
    };

    const int Cins[3] = {256, 512, 1024};
    PKH k2a[3], k2b[3], k2c[3], k3b[3], k4a[3], k4b[3], k4c[3], k4d[3];
    for (int i = 0; i < 3; ++i) {
        k2a[i] = packBN(cv2a[i],  64, Cins[i], 9,  64);
        k2b[i] = packBN(cv2b[i],  64, 64,      9,  64);
        k2c[i] = packBias(cv2c[i], 64, 64,     1,  64);
        k3b[i] = packBN(cv3b[i], 128, Cins[i], 1, 128);
        k4a[i] = packBN(cv4a[i], 128, Cins[i], 9, 128);
        k4b[i] = packBN(cv4b[i], 128, 128,     9, 128);
        k4c[i] = packBN(cv4c3[i],128, 128,     9, 128);
        k4d[i] = packBias(cv4d[i], 63, 128,    1,  64);
    }

    _Float16* XH = (_Float16*)(ws + OFF_XH);
    _Float16* DW = (_Float16*)(ws + OFF_DW);
    _Float16* T1 = (_Float16*)(ws + OFF_T1);
    _Float16* T2 = (_Float16*)(ws + OFF_T2);
    float* DET = (float*)(ws + OFF_DET);
    float* KPT = (float*)(ws + OFF_KPT);
    int*   SEL = (int*)(ws + OFF_SEL);

    const int Hs[3]    = {80, 40, 20};
    const int aoffs[3] = {0, 6400, 8000};

    auto conv = [&](const _Float16* Xl, const PKH& pk, int H, int W, int Cin, int taps,
                    int mode, _Float16* Yh, float* Yf, int CoutStore,
                    int bstride, int coff, int aoff) {
        int HW = H * W, Ptot = 16 * HW;
        dim3 grid((Ptot + 127) / 128, pk.Opad / 64);
        conv_wmma<<<grid, 256, 0, stream>>>(Xl, pk.wt, pk.sc, pk.bi, Yh, Yf,
                                            H, W, Cin, taps, CoutStore,
                                            HW, Ptot, bstride, coff, aoff, mode);
    };

    for (int i = 0; i < 3; ++i) {
        int H = Hs[i], W = H, HW = H * W, Ptot = 16 * HW, Cin = Cins[i], aoff = aoffs[i];
        nchw_to_nhwc_f16<<<(Ptot * Cin + 255) / 256, 256, 0, stream>>>(Xin[i], XH, Cin, HW, Ptot);
        // cv2: box-regression branch
        conv(XH, k2a[i], H, W, Cin, 9, 0, T1, nullptr, 64, 0, 0, 0);
        conv(T1, k2b[i], H, W, 64,  9, 0, T2, nullptr, 64, 0, 0, 0);
        conv(T2, k2c[i], H, W, 64,  1, 1, nullptr, DET, 64, 65 * 8400, 0, aoff);
        // cv4: keypoint branch
        conv(XH, k4a[i], H, W, Cin, 9, 0, T1, nullptr, 128, 0, 0, 0);
        conv(T1, k4b[i], H, W, 128, 9, 0, T2, nullptr, 128, 0, 0, 0);
        conv(T2, k4c[i], H, W, 128, 9, 0, T1, nullptr, 128, 0, 0, 0);
        conv(T1, k4d[i], H, W, 128, 1, 1, nullptr, KPT, 63, 63 * 8400, 0, aoff);
        // cv3: class branch (depthwise + 1x1 CBS + 1x1 cls)
        dw_cbs<<<(Ptot * Cin + 255) / 256, 256, 0, stream>>>(
            XH, cv3a[i].w, cv3a[i].g, cv3a[i].be, cv3a[i].me, cv3a[i].va,
            DW, Cin, H, W, HW, Ptot);
        conv(DW, k3b[i], H, W, Cin, 1, 0, T1, nullptr, 128, 0, 0, 0);
        cls_head<<<(Ptot + 255) / 256, 256, 0, stream>>>(T1, cv3c[i].w, cv3c[i].b,
                                                         DET, HW, Ptot, aoff);
    }

    select_best<<<16, 256, 0, stream>>>(DET, KPT, SEL);
    emit_out<<<16, 96, 0, stream>>>(DET, KPT, SEL, (float*)d_out);
}
